// VanillaRNN_62534723829995
// MI455X (gfx1250) — compile-verified, matrix operand order checked by reading, not run on hardware
//
#include <hip/hip_runtime.h>

#define S_LEN 1024
#define HID   4096

typedef __attribute__((ext_vector_type(16))) __bf16 v16bf;
typedef __attribute__((ext_vector_type(8)))  __bf16 v8bf;
typedef __attribute__((ext_vector_type(4)))  __bf16 v4bf;
typedef __attribute__((ext_vector_type(8)))  float  v8f;
typedef int v4i __attribute__((vector_size(16)));   // matches builtin param pointee

#if __has_builtin(__builtin_amdgcn_global_load_async_to_lds_b128)
#define HAVE_ASYNC_LDS 1
#else
#define HAVE_ASYNC_LDS 0
#endif

__device__ __forceinline__ void wait_async0() {
#if __has_builtin(__builtin_amdgcn_s_wait_asynccnt)
    __builtin_amdgcn_s_wait_asynccnt(0);
#else
    asm volatile("s_wait_asynccnt 0x0" ::: "memory");
#endif
}

// Copy 16 bytes (8 bf16) global -> LDS via the async engine (ASYNCcnt-tracked).
// Signature (probe-confirmed): (global v4i* src, lds v4i* dst, imm offset, imm cpol)
__device__ __forceinline__ void cp16_g2l(__bf16* lds_dst, const __bf16* gsrc) {
#if HAVE_ASYNC_LDS
    __builtin_amdgcn_global_load_async_to_lds_b128(
        (v4i*)gsrc, (v4i*)lds_dst, /*offset=*/0, /*cpol=*/0);
#else
    *(v8bf*)lds_dst = *(const v8bf*)gsrc;
#endif
}

// ---------------- elementwise f32 -> bf16 conversion ----------------
__global__ __launch_bounds__(256)
void cvt_f32_bf16(const float* __restrict__ src, __bf16* __restrict__ dst, int n4) {
    int i = blockIdx.x * blockDim.x + threadIdx.x;
    if (i < n4) {
        float4 v = ((const float4*)src)[i];
        v4bf o;
        o.x = (__bf16)v.x; o.y = (__bf16)v.y; o.z = (__bf16)v.z; o.w = (__bf16)v.w;
        ((v4bf*)dst)[i] = o;
    }
}

__global__ __launch_bounds__(256)
void zero_f32(float* __restrict__ p, int n) {
    int i = blockIdx.x * blockDim.x + threadIdx.x;
    if (i < n) p[i] = 0.0f;
}

// ---------------- WMMA GEMM: xp[m][n] = sum_k x_bf[m][k] * whx_bf[n][k] + b_h[n] ----------------
// w_hx used in native [n][k] layout (xp = x @ w_hx.T needs B^T[n][k] = w_hx[n][k]).
// Block: 256 threads = 8 wave32s. Block tile 128(M) x 128(N), BK = 32.
// Double-buffered LDS staging via async global->LDS copies, overlapped with WMMA.
__global__ __launch_bounds__(256)
void gemm_xp(const __bf16* __restrict__ A,    // [S][H]  (x in bf16)
             const __bf16* __restrict__ B,    // [N][K]  (w_hx in bf16, native layout)
             const float*  __restrict__ bias, // b_h [H]
             float*        __restrict__ C)    // xp [S][H]
{
    __shared__ __bf16 As[2][128 * 32];   // [m][k], k innermost
    __shared__ __bf16 Bs[2][128 * 32];   // [n][k], k innermost

    const int tid  = threadIdx.x;
    const int lane = tid & 31;
    const int wave = tid >> 5;
    const int n0 = blockIdx.x * 128;
    const int m0 = blockIdx.y * 128;

    v8f acc[8] = {};

    // Per-lane fragment addressing for 16-bit A/B operands (ISA 7.12.2):
    // lanes 0-15  : M/N = lane,    elems 0-7 = K+0..7,  elems 8-15 = K+16..23
    // lanes 16-31 : M/N = lane-16, elems 0-7 = K+8..15, elems 8-15 = K+24..31
    const int ml = lane & 15;
    const int kb = (lane >> 4) << 3;           // 0 or 8
    const int aoff = (wave * 16 + ml) * 32 + kb;
    const int boff = ml * 32 + kb;

    // Stage one 128x32 bf16 tile pair (2 x 8 KB = 2 x 512 x 16B chunks; 2 chunks/thread/tile).
    auto stage = [&](int buf, int k0) {
        #pragma unroll
        for (int j = 0; j < 2; ++j) {
            int chunk = tid + j * 256;        // 0..511
            int r = chunk >> 2;               // tile row
            int c = (chunk & 3) * 8;          // bf16 element offset within row
            cp16_g2l(&As[buf][r * 32 + c], A + (size_t)(m0 + r) * HID + k0 + c);
            cp16_g2l(&Bs[buf][r * 32 + c], B + (size_t)(n0 + r) * HID + k0 + c);
        }
    };

    stage(0, 0);
    int cur = 0;
    #pragma clang loop unroll(disable)
    for (int kt = 0; kt < HID / 32; ++kt) {
        wait_async0();           // my async writes for 'cur' have landed in LDS
        __syncthreads();         // everyone's writes visible; everyone done reading buf cur^1
        if (kt + 1 < HID / 32)
            stage(cur ^ 1, (kt + 1) * 32);   // prefetch next K-step into the other buffer

        const __bf16* abase = &As[cur][aoff];
        const __bf16* bbase = &Bs[cur][boff];

        // A fragment: two contiguous 16B LDS reads
        v8bf a0 = *(const v8bf*)(abase);
        v8bf a1 = *(const v8bf*)(abase + 16);
        v16bf av = __builtin_shufflevector(a0, a1, 0,1,2,3,4,5,6,7,8,9,10,11,12,13,14,15);

        #pragma unroll
        for (int nt = 0; nt < 8; ++nt) {
            const __bf16* bp = bbase + nt * (16 * 32);
            v8bf b0 = *(const v8bf*)(bp);
            v8bf b1 = *(const v8bf*)(bp + 16);
            v16bf bv = __builtin_shufflevector(b0, b1, 0,1,2,3,4,5,6,7,8,9,10,11,12,13,14,15);
            acc[nt] = __builtin_amdgcn_wmma_f32_16x16x32_bf16(
                /*neg_a=*/false, av, /*neg_b=*/false, bv,
                /*c_mod=*/(short)0, acc[nt], /*reuse_a=*/false, /*reuse_b=*/false);
        }
        cur ^= 1;
    }

    // Epilogue: C/D layout (ISA 7.12.2): VGPR r -> M = r (lanes 0-15) or 8+r (lanes 16-31)
    const int rsel = (lane >> 4) << 3;
    #pragma unroll
    for (int nt = 0; nt < 8; ++nt) {
        int gc = n0 + nt * 16 + ml;
        float bv = bias[gc];
        #pragma unroll
        for (int r = 0; r < 8; ++r) {
            int gr = m0 + wave * 16 + rsel + r;
            C[(size_t)gr * HID + gc] = acc[nt][r] + bv;
        }
    }
}

// ---------------- one recurrence step: h_out = tanh(xp_t + W_hh @ h_in) ----------------
// Wave per output row; W_hh in bf16 (32 MB -> resident in 192 MB L2 across all 1024 steps).
__global__ __launch_bounds__(256)
void rnn_step(const __bf16* __restrict__ Whh, const float* __restrict__ xp_row,
              const float* __restrict__ h_in, float* __restrict__ h_out)
{
    const int lane = threadIdx.x & 31;
    const int wave = threadIdx.x >> 5;
    const int row  = blockIdx.x * 8 + wave;

    const unsigned int* w2 = (const unsigned int*)(Whh + (size_t)row * HID); // bf16x2
    const float2*       h2 = (const float2*)h_in;

    float acc = 0.0f;
    #pragma unroll 8
    for (int j = 0; j < HID / 64; ++j) {       // 64 packed pairs per lane
        int p = lane + 32 * j;
        unsigned int w = w2[p];
        float2 hv = h2[p];
        float b0 = (float)__builtin_bit_cast(__bf16, (unsigned short)(w & 0xffffu));
        float b1 = (float)__builtin_bit_cast(__bf16, (unsigned short)(w >> 16));
        acc = fmaf(b0, hv.x, acc);
        acc = fmaf(b1, hv.y, acc);
    }
    #pragma unroll
    for (int off = 16; off > 0; off >>= 1)
        acc += __shfl_xor(acc, off, 32);
    if (lane == 0)
        h_out[row] = tanhf(acc + xp_row[row]);
}

// ---------------- final projection: out = w_ph @ h + b_p (f32) ----------------
__global__ __launch_bounds__(256)
void final_proj(const float* __restrict__ Wph, const float* __restrict__ h,
                const float* __restrict__ bp, float* __restrict__ out)
{
    const int lane = threadIdx.x & 31;
    const int wave = threadIdx.x >> 5;
    const int row  = blockIdx.x * 8 + wave;
    const float* wr = Wph + (size_t)row * HID;

    float acc = 0.0f;
    #pragma unroll 8
    for (int j = 0; j < HID / 32; ++j)
        acc = fmaf(wr[lane + 32 * j], h[lane + 32 * j], acc);
    #pragma unroll
    for (int off = 16; off > 0; off >>= 1)
        acc += __shfl_xor(acc, off, 32);
    if (lane == 0)
        out[row] = acc + bp[row];
}

extern "C" void kernel_launch(void* const* d_in, const int* in_sizes, int n_in,
                              void* d_out, int out_size, void* d_ws, size_t ws_size,
                              hipStream_t stream)
{
    (void)in_sizes; (void)n_in; (void)out_size; (void)ws_size;

    const float* x    = (const float*)d_in[0];
    const float* w_hx = (const float*)d_in[1];
    const float* w_hh = (const float*)d_in[2];
    const float* b_h  = (const float*)d_in[3];
    const float* w_ph = (const float*)d_in[4];
    const float* b_p  = (const float*)d_in[5];

    char* ws = (char*)d_ws;
    __bf16* xbf  = (__bf16*)(ws);                                  //  8 MB  [S][H]
    __bf16* whxb = (__bf16*)(ws + (size_t)8  * 1024 * 1024);       // 32 MB  [N][K] (native layout)
    __bf16* whhb = (__bf16*)(ws + (size_t)40 * 1024 * 1024);       // 32 MB  [H][H]
    float*  xp   = (float*) (ws + (size_t)72 * 1024 * 1024);       // 16 MB  [S][H]
    float*  hbuf = (float*) (ws + (size_t)88 * 1024 * 1024);       // 32 KB  ping-pong h

    // Precision conversions (no transpose needed: w_hx rows are already K-innermost)
    cvt_f32_bf16<<<(S_LEN * HID / 4 + 255) / 256, 256, 0, stream>>>(x, xbf, S_LEN * HID / 4);
    cvt_f32_bf16<<<(HID * HID / 4 + 255) / 256, 256, 0, stream>>>(w_hx, whxb, HID * HID / 4);
    cvt_f32_bf16<<<(HID * HID / 4 + 255) / 256, 256, 0, stream>>>(w_hh, whhb, HID * HID / 4);
    zero_f32<<<(2 * HID) / 256, 256, 0, stream>>>(hbuf, 2 * HID);

    // Input projection GEMM (WMMA bf16, async double-buffered LDS): xp = x @ w_hx^T + b_h
    gemm_xp<<<dim3(HID / 128, S_LEN / 128), 256, 0, stream>>>(xbf, whxb, b_h, xp);

    // Sequential recurrence: 1024 dependent GEMVs, W_hh streamed from L2
    for (int t = 0; t < S_LEN; ++t) {
        const float* hin  = hbuf + (size_t)(t & 1) * HID;
        float*       hout = hbuf + (size_t)((t + 1) & 1) * HID;
        rnn_step<<<HID / 8, 256, 0, stream>>>(whhb, xp + (size_t)t * HID, hin, hout);
    }

    // Final projection to d_out (f32, 4096 elements)
    final_proj<<<HID / 8, 256, 0, stream>>>(w_ph, hbuf, b_p, (float*)d_out);
}
